// Attention_50749333569807
// MI455X (gfx1250) — matrix operand order repaired and checked by reading, not compiled, not run on hardware
//
#include <hip/hip_runtime.h>
#include <hip/hip_bf16.h>

// ---------------------------------------------------------------------------
// Types for CDNA5 WMMA (wave32): v16bf A/B fragments, v8f accumulator.
// ---------------------------------------------------------------------------
typedef __attribute__((ext_vector_type(16))) __bf16 v16bf;
typedef __attribute__((ext_vector_type(8)))  __bf16 v8bf;
typedef __attribute__((ext_vector_type(2)))  __bf16 v2bf;
typedef __attribute__((ext_vector_type(8)))  float  v8f;
typedef __attribute__((ext_vector_type(4)))  unsigned int u32x4;
typedef __attribute__((ext_vector_type(8)))  int i32x8;
typedef __attribute__((ext_vector_type(4)))  int i32x4;

#define B_      4
#define DIM_    384
#define HEADS_  8
#define CH_     48      // DIM_/HEADS_
#define HW_     128
#define NSP_    16384   // HW_*HW_
#define C3_     1152    // 3*DIM_

// Native bf16 conversion (lets clang use v_cvt_pk_bf16_f32).
__device__ __forceinline__ unsigned pack2(float a, float b) {
    v2bf v; v[0] = (__bf16)a; v[1] = (__bf16)b;
    return __builtin_bit_cast(unsigned, v);
}
__device__ __forceinline__ unsigned short bf16s(float f) {
    return __builtin_bit_cast(unsigned short, (__bf16)f);
}
__device__ __forceinline__ unsigned lds_off(const void* p) {
    // LDS aperture maps addr[31:0] to the in-workgroup LDS offset.
    return (unsigned)(unsigned long long)p;
}

__device__ __forceinline__ v8f wmma_bf16(v16bf a, v16bf b, v8f c) {
    // v_wmma_f32_16x16x32_bf16  (neg_a, A, neg_b, B, c_mod, C, reuse_a, reuse_b)
    return __builtin_amdgcn_wmma_f32_16x16x32_bf16(false, a, false, b,
                                                   (short)0, c, false, false);
}

// A fragment (16x32 bf16, MxK), ISA 7.12.2 layout:
// lane L: M = L%16; VGPRs 0-3 hold K = base..base+7 (base = (L/16)*8),
// VGPRs 4-7 hold K = 16+base..16+base+7. Two contiguous 16B LDS loads.
template<int STRIDE>
__device__ __forceinline__ v16bf load_a_frag(const unsigned short* base,
                                             int row0, int col0, int lane) {
    int row = row0 + (lane & 15);
    int cb  = col0 + ((lane >> 4) << 3);
    const unsigned short* p = base + row * STRIDE + cb;
    v8bf lo = *(const v8bf*)p;
    v8bf hi = *(const v8bf*)(p + 16);
    v16bf r;
#pragma unroll
    for (int i = 0; i < 8; ++i) { r[i] = lo[i]; r[i + 8] = hi[i]; }
    return r;
}

// B fragment (32x16 bf16, KxN) read from an LDS tile stored TRANSPOSED
// as [N][K]: lane L: N = L%16, K = (L/16)*8 + i (i<8), 16 + (L/16)*8 + (i-8) (i>=8).
template<int STRIDE>
__device__ __forceinline__ v16bf load_b_frag(const unsigned short* base,
                                             int ncol0, int k0, int lane) {
    int n  = ncol0 + (lane & 15);
    int kb = k0 + ((lane >> 4) << 3);
    const unsigned short* p = base + n * STRIDE + kb;
    v8bf lo = *(const v8bf*)p;
    v8bf hi = *(const v8bf*)(p + 16);
    v16bf r;
#pragma unroll
    for (int i = 0; i < 8; ++i) { r[i] = lo[i]; r[i + 8] = hi[i]; }
    return r;
}

// ---------------------------------------------------------------------------
// Kernel 1/6: Y[m,n] = sum_k W[m,k] * X[k,n] per batch (1x1 conv GEMM).
// Block computes 128(M) x 128(N); 8 waves in 4x2 grid, each 32M x 64N.
// LDS double-buffered: next K-tile global loads overlap current WMMAs.
// ---------------------------------------------------------------------------
__global__ __launch_bounds__(256)
void gemm_wmma_bf16(const float* __restrict__ W, const float* __restrict__ X,
                    float* __restrict__ Y, int M, int K) {
    __shared__ __align__(16) unsigned short As[2][128 * 32]; // [m][k]
    __shared__ __align__(16) unsigned short Bs[2][128 * 32]; // [n][k] transposed

    const int t    = threadIdx.x;
    const int lane = t & 31;
    const int wave = t >> 5;
    const int m0   = blockIdx.y * 128;
    const int n0   = blockIdx.x * 128;
    const size_t bofX = (size_t)blockIdx.z * (size_t)K * NSP_;
    const size_t bofY = (size_t)blockIdx.z * (size_t)M * NSP_;

    const int mw = (wave >> 1) * 32; // 0,32,64,96
    const int nw = (wave & 1) * 64;  // 0,64

    const int rowA  = t >> 1;        // 0..127
    const int halfA = (t & 1) * 16;  // col offset 0/16
    const int krB   = t >> 3;        // 0..31
    const int nsegB = (t & 7) * 16;  // 0..112

    auto stageA = [&](int k0, int buf) {
        const float* wp = W + (size_t)(m0 + rowA) * K + k0 + halfA;
        if (k0 + 32 < K) __builtin_prefetch(wp + 32, 0, 3);
        unsigned* dst = (unsigned*)&As[buf][rowA * 32 + halfA];
#pragma unroll
        for (int j = 0; j < 4; ++j) {
            float4 f = ((const float4*)wp)[j];
            dst[j * 2 + 0] = pack2(f.x, f.y);
            dst[j * 2 + 1] = pack2(f.z, f.w);
        }
    };
    auto stageB = [&](int k0, int buf) {
        const float* xp = X + bofX + (size_t)(k0 + krB) * NSP_ + n0 + nsegB;
        if (k0 + 32 < K) __builtin_prefetch(xp + 32 * NSP_, 0, 3);
        unsigned short* bb = Bs[buf];
#pragma unroll
        for (int j = 0; j < 4; ++j) {
            float4 f = ((const float4*)xp)[j];
            int nc = nsegB + j * 4;
            bb[(nc + 0) * 32 + krB] = bf16s(f.x);
            bb[(nc + 1) * 32 + krB] = bf16s(f.y);
            bb[(nc + 2) * 32 + krB] = bf16s(f.z);
            bb[(nc + 3) * 32 + krB] = bf16s(f.w);
        }
    };

    v8f zero = {0.f, 0.f, 0.f, 0.f, 0.f, 0.f, 0.f, 0.f};
    v8f acc[2][4];
#pragma unroll
    for (int i = 0; i < 2; ++i)
#pragma unroll
        for (int j = 0; j < 4; ++j) acc[i][j] = zero;

    const int nk = K / 32;
    stageA(0, 0);
    stageB(0, 0);
    __syncthreads();

    for (int kt = 0; kt < nk; ++kt) {
        const int cur = kt & 1;
        if (kt + 1 < nk) {            // overlap next-tile loads with WMMAs
            stageA((kt + 1) * 32, cur ^ 1);
            stageB((kt + 1) * 32, cur ^ 1);
        }
        v16bf a0 = load_a_frag<32>(As[cur], mw, 0, lane);
        v16bf a1 = load_a_frag<32>(As[cur], mw + 16, 0, lane);
#pragma unroll
        for (int nt = 0; nt < 4; ++nt) {
            v16bf bb = load_b_frag<32>(Bs[cur], nw + nt * 16, 0, lane);
            acc[0][nt] = wmma_bf16(a0, bb, acc[0][nt]);
            acc[1][nt] = wmma_bf16(a1, bb, acc[1][nt]);
        }
        __syncthreads();
    }

    const int mb = m0 + mw + 8 * (lane >> 4);
    const int nb = n0 + nw + (lane & 15);
#pragma unroll
    for (int mt = 0; mt < 2; ++mt)
#pragma unroll
        for (int nt = 0; nt < 4; ++nt)
#pragma unroll
            for (int r = 0; r < 8; ++r)
                Y[bofY + (size_t)(mb + mt * 16 + r) * NSP_ + nb + nt * 16] =
                    acc[mt][nt][r];
}

// ---------------------------------------------------------------------------
// Kernel 2: depthwise 3x3, pad 1.  One block per (b, channel).
// Whole 128x128 fp32 image (64KB) DMA'd into LDS by the Tensor Data Mover
// (one TENSOR_LOAD_TO_LDS with a 2D D#), drained with s_wait_tensorcnt.
// ---------------------------------------------------------------------------
__global__ __launch_bounds__(256)
void dwconv3x3(const float* __restrict__ in, const float* __restrict__ wdw,
               float* __restrict__ out) {
    __shared__ __align__(16) float img[NSP_];   // 64KB
    const int bc = blockIdx.x;                  // b*1152 + ch
    const int ch = bc % C3_;
    const float* src = in + (size_t)bc * NSP_;
    float* dst = out + (size_t)bc * NSP_;

    float w[9];
#pragma unroll
    for (int i = 0; i < 9; ++i) w[i] = wdw[ch * 9 + i];

    if (threadIdx.x < 32) {  // one wave issues the TDM op (TDM ignores EXEC)
        unsigned long long ga = (unsigned long long)src;
        // D# group 0: count=1 (valid), lds_addr, 57-bit global_addr, type=2.
        u32x4 g0;
        g0[0] = 1u;
        g0[1] = lds_off(&img[0]);
        g0[2] = (unsigned)(ga & 0xFFFFFFFFull);
        g0[3] = (unsigned)((ga >> 32) & 0x1FFFFFFull) | (2u << 30);
        // D# group 1: mask=0 (no cluster), data_size=4B, tensor 128x128,
        // tile 128x128, dim0 stride = 128 elements.
        i32x8 g1;
        g1[0] = (2 << 16);          // data_size = 2 -> 4 bytes
        g1[1] = (int)(128u << 16);  // tensor_dim0[15:0] = 128
        g1[2] = (int)(128u << 16);  // tensor_dim0 hi=0 | tensor_dim1[15:0]=128
        g1[3] = (int)(128u << 16);  // tensor_dim1 hi=0 | tile_dim0=128
        g1[4] = 128;                // tile_dim1=128, tile_dim2=0
        g1[5] = 128;                // tensor_dim0_stride = 128
        g1[6] = 0;
        g1[7] = 0;
        i32x4 gz4 = {0, 0, 0, 0};           // groups 2/3 unused (2D tensor)
        i32x8 gz8 = {0, 0, 0, 0, 0, 0, 0, 0};
        __builtin_amdgcn_tensor_load_to_lds(g0, g1, gz4, gz4, gz8, 0);
        __builtin_amdgcn_s_wait_tensorcnt(0);
    }
    __syncthreads();

    for (int p = threadIdx.x; p < NSP_; p += 256) {
        int y = p >> 7, x = p & 127;
        float s = 0.f;
#pragma unroll
        for (int dy = -1; dy <= 1; ++dy) {
            int yy = y + dy;
            if (yy < 0 || yy > 127) continue;
#pragma unroll
            for (int dx = -1; dx <= 1; ++dx) {
                int xx = x + dx;
                if (xx < 0 || xx > 127) continue;
                s += w[(dy + 1) * 3 + (dx + 1)] * img[yy * HW_ + xx];
            }
        }
        dst[p] = s;
    }
}

// ---------------------------------------------------------------------------
// Kernel 3: sum of squares per (b, channel<768) for q/k L2 normalize.
// Per-wave double-buffered async-to-LDS streaming (ASYNCcnt pipeline):
// chunk c+1 issues while chunk c is reduced.
// ---------------------------------------------------------------------------
__global__ __launch_bounds__(256)
void rownorm(const float* __restrict__ qkvd, float* __restrict__ norms) {
    __shared__ __align__(16) float buf[8][2][128];  // 8KB, per-wave dbl buffer
    const int blk = blockIdx.x;                     // b*768 + c
    const int b = blk / 768, c = blk % 768;
    const float* src = qkvd + ((size_t)b * C3_ + c) * NSP_;

    const int t = threadIdx.x, lane = t & 31, wave = t >> 5;
    const int base = wave * 2048;                   // per-wave slice

    auto stage = [&](int chunk, int sb) {
        unsigned ldsa = lds_off(&buf[wave][sb][lane * 4]);
        unsigned long long ga =
            (unsigned long long)(src + base + chunk * 128 + lane * 4);
        asm volatile("global_load_async_to_lds_b128 %0, %1, off"
                     :: "v"(ldsa), "v"(ga) : "memory");
    };

    float s = 0.f;
    stage(0, 0);
#pragma unroll
    for (int ck = 0; ck < 16; ++ck) {
        if (ck + 1 < 16) {
            stage(ck + 1, (ck + 1) & 1);
            asm volatile("s_wait_asynccnt 1" ::: "memory");
        } else {
            asm volatile("s_wait_asynccnt 0" ::: "memory");
        }
        float4 f = *(const float4*)&buf[wave][ck & 1][lane * 4];
        s += f.x * f.x + f.y * f.y + f.z * f.z + f.w * f.w;
    }

    __shared__ float red[256];
    red[t] = s;
    __syncthreads();
    for (int off = 128; off > 0; off >>= 1) {
        if (t < off) red[t] += red[t + off];
        __syncthreads();
    }
    if (t == 0) norms[blk] = red[0];
}

// ---------------------------------------------------------------------------
// Kernel 4: attn = softmax( (q/|q|) @ (k/|k|)^T * temp )  per (b, head).
// One block per (b,head); 8 waves split K=16384; WMMA 3x3 tiles of 16x16.
// ---------------------------------------------------------------------------
__global__ __launch_bounds__(256)
void attn_qk(const float* __restrict__ qkvd, const float* __restrict__ norms,
             const float* __restrict__ temp, float* __restrict__ attn) {
    const int bh = blockIdx.x;
    const int b = bh >> 3, h = bh & 7;

    __shared__ __align__(16) unsigned short tile[8][96 * 32]; // per-wave [ch][n]
    __shared__ float attnS[48 * 48];
    __shared__ float scaleS[96];

    const int t = threadIdx.x, lane = t & 31, wave = t >> 5;

    if (t < 96) {
        int gch = (t < 48) ? (h * CH_ + t) : (DIM_ + h * CH_ + (t - 48));
        float n2 = norms[b * 768 + gch];
        scaleS[t] = 1.f / fmaxf(sqrtf(n2), 1e-12f);
    }
    for (int i = t; i < 48 * 48; i += 256) attnS[i] = 0.f;
    __syncthreads();

    unsigned short* tw = tile[wave];
    const int nb = wave * 2048;
    const int r0 = lane * 3;

    v8f zero = {0.f, 0.f, 0.f, 0.f, 0.f, 0.f, 0.f, 0.f};
    v8f acc[3][3];
#pragma unroll
    for (int i = 0; i < 3; ++i)
#pragma unroll
        for (int j = 0; j < 3; ++j) acc[i][j] = zero;

    for (int it = 0; it < 64; ++it) {
        const int n0 = nb + it * 32;
#pragma unroll
        for (int rr = 0; rr < 3; ++rr) {
            int r = r0 + rr;
            int gch = (r < 48) ? (h * CH_ + r) : (DIM_ + h * CH_ + (r - 48));
            const float* src = qkvd + ((size_t)b * C3_ + gch) * NSP_ + n0;
            float sc = scaleS[r];
            unsigned* dst = (unsigned*)&tw[r * 32];
#pragma unroll
            for (int j = 0; j < 8; ++j) {
                float4 f = ((const float4*)src)[j];
                dst[j * 2 + 0] = pack2(f.x * sc, f.y * sc);
                dst[j * 2 + 1] = pack2(f.z * sc, f.w * sc);
            }
        }
        __syncthreads();
        v16bf aq[3];
#pragma unroll
        for (int mt = 0; mt < 3; ++mt) aq[mt] = load_a_frag<32>(tw, mt * 16, 0, lane);
#pragma unroll
        for (int nt = 0; nt < 3; ++nt) {
            v16bf bk = load_b_frag<32>(tw, 48 + nt * 16, 0, lane);
#pragma unroll
            for (int mt = 0; mt < 3; ++mt)
                acc[mt][nt] = wmma_bf16(aq[mt], bk, acc[mt][nt]);
        }
        __syncthreads();
    }

    // Reduce partial 48x48 across waves via LDS float atomics.
    const int mb = 8 * (lane >> 4);
    const int db = lane & 15;
#pragma unroll
    for (int mt = 0; mt < 3; ++mt)
#pragma unroll
        for (int nt = 0; nt < 3; ++nt)
#pragma unroll
            for (int r = 0; r < 8; ++r)
                atomicAdd(&attnS[(mt * 16 + r + mb) * 48 + nt * 16 + db],
                          acc[mt][nt][r]);
    __syncthreads();

    // Temperature scale + row softmax; one thread per row of 48.
    if (t < 48) {
        float tp = temp[h];
        float* row = &attnS[t * 48];
        float mx = -1e30f;
        for (int d = 0; d < 48; ++d) { row[d] *= tp; mx = fmaxf(mx, row[d]); }
        float s = 0.f;
        for (int d = 0; d < 48; ++d) { float e = expf(row[d] - mx); row[d] = e; s += e; }
        float inv = 1.f / s;
        float* out = attn + ((size_t)bh * 48 + t) * 48;
        for (int d = 0; d < 48; ++d) out[d] = row[d] * inv;
    }
}

// ---------------------------------------------------------------------------
// Kernel 5: out = attn @ v  per (b, head).  M=48, K=48 (padded to 64), N tile 256.
// ---------------------------------------------------------------------------
__global__ __launch_bounds__(256)
void attn_v(const float* __restrict__ qkvd, const float* __restrict__ attn,
            float* __restrict__ outb) {
    const int bh = blockIdx.y;
    const int b = bh >> 3, h = bh & 7;
    const int n0 = blockIdx.x * 256;

    __shared__ __align__(16) unsigned short Aat[48 * 64];  // attn, K padded
    __shared__ __align__(16) unsigned short Vs[256 * 64];  // [n][d] transposed

    const int t = threadIdx.x, lane = t & 31, wave = t >> 5;

    for (int idx = t; idx < 48 * 64; idx += 256) {
        int c = idx >> 6, d = idx & 63;
        float v = (d < 48) ? attn[((size_t)bh * 48 + c) * 48 + d] : 0.f;
        Aat[idx] = bf16s(v);
    }
    {
        int d = t >> 2;               // 0..63
        int ns = (t & 3) * 64;        // 0..192
        if (d < 48) {
            const float* src =
                qkvd + ((size_t)b * C3_ + 2 * DIM_ + h * CH_ + d) * NSP_ + n0 + ns;
            for (int j = 0; j < 64; j += 4) {
                float4 f = *(const float4*)(src + j);
                Vs[(ns + j + 0) * 64 + d] = bf16s(f.x);
                Vs[(ns + j + 1) * 64 + d] = bf16s(f.y);
                Vs[(ns + j + 2) * 64 + d] = bf16s(f.z);
                Vs[(ns + j + 3) * 64 + d] = bf16s(f.w);
            }
        } else {
            for (int j = 0; j < 64; ++j) Vs[(ns + j) * 64 + d] = 0;
        }
    }
    __syncthreads();

    v8f zero = {0.f, 0.f, 0.f, 0.f, 0.f, 0.f, 0.f, 0.f};
    v8f acc[3][2];
#pragma unroll
    for (int i = 0; i < 3; ++i) { acc[i][0] = zero; acc[i][1] = zero; }

    const int nw = wave * 32;
#pragma unroll
    for (int k0 = 0; k0 < 64; k0 += 32) {
        v16bf a[3];
#pragma unroll
        for (int mt = 0; mt < 3; ++mt) a[mt] = load_a_frag<64>(Aat, mt * 16, k0, lane);
#pragma unroll
        for (int nt = 0; nt < 2; ++nt) {
            v16bf bb = load_b_frag<64>(Vs, nw + nt * 16, k0, lane);
#pragma unroll
            for (int mt = 0; mt < 3; ++mt)
                acc[mt][nt] = wmma_bf16(a[mt], bb, acc[mt][nt]);
        }
    }

    const int mb = 8 * (lane >> 4);
    const int nb2 = n0 + nw + (lane & 15);
#pragma unroll
    for (int mt = 0; mt < 3; ++mt)
#pragma unroll
        for (int nt = 0; nt < 2; ++nt)
#pragma unroll
            for (int r = 0; r < 8; ++r) {
                int m = mt * 16 + r + mb; // 0..47
                outb[((size_t)b * DIM_ + h * CH_ + m) * NSP_ + nb2 + nt * 16] =
                    acc[mt][nt][r];
            }
}

// ---------------------------------------------------------------------------
extern "C" void kernel_launch(void* const* d_in, const int* in_sizes, int n_in,
                              void* d_out, int out_size, void* d_ws, size_t ws_size,
                              hipStream_t stream) {
    const float* x      = (const float*)d_in[0];
    const float* w_qkv  = (const float*)d_in[1];
    const float* w_dw   = (const float*)d_in[2];
    const float* w_proj = (const float*)d_in[3];
    const float* temp   = (const float*)d_in[4];

    const size_t qkvElems = (size_t)B_ * C3_ * NSP_; // 75,497,472 floats
    float* qkv   = (float*)d_ws;
    float* qkvd  = qkv + qkvElems;
    float* norms = qkvd + qkvElems;                   // B*768
    float* attn  = norms + (size_t)B_ * 768;          // B*8*48*48
    float* outb  = qkv;  // reuse: qkv dead after dwconv/rownorm

    dim3 blk(256);

    // 1) qkv = w_qkv @ x          (M=1152, K=384)
    gemm_wmma_bf16<<<dim3(NSP_ / 128, C3_ / 128, B_), blk, 0, stream>>>(
        w_qkv, x, qkv, C3_, DIM_);
    // 2) depthwise 3x3 (TDM-staged image in LDS)
    dwconv3x3<<<dim3(B_ * C3_), blk, 0, stream>>>(qkv, w_dw, qkvd);
    // 3) q,k row norms (async-to-LDS double-buffered stream)
    rownorm<<<dim3(B_ * 768), blk, 0, stream>>>(qkvd, norms);
    // 4) attn = softmax(qn @ kn^T * temp)
    attn_qk<<<dim3(B_ * HEADS_), blk, 0, stream>>>(qkvd, norms, temp, attn);
    // 5) out = attn @ v
    attn_v<<<dim3(NSP_ / 256, B_ * HEADS_), blk, 0, stream>>>(qkvd, attn, outb);
    // 6) final = w_proj @ out     (M=384, K=384)
    gemm_wmma_bf16<<<dim3(NSP_ / 128, DIM_ / 128, B_), blk, 0, stream>>>(
        w_proj, outb, (float*)d_out, DIM_, DIM_);
}
